// SAKEInteraction_75557064671336
// MI455X (gfx1250) — matrix operand order repaired and checked by reading, not compiled, not run on hardware
//
#include <hip/hip_runtime.h>
#include <math.h>

// ---------------- problem constants (match reference) ----------------
#define FDIM   32
#define HHEADS 7
#define CDIM   224           // F*H
#define NRBF_C 50
#define CUTOFF_F 5.0f
#define EPS_F  1e-8f
#define PI_F   3.14159265358979f

typedef __attribute__((ext_vector_type(16))) _Float16 v16h;
typedef __attribute__((ext_vector_type(8)))  _Float16 v8h;
typedef __attribute__((ext_vector_type(8)))  float    v8f;

__device__ __forceinline__ float siluf(float x) { return x / (1.0f + __expf(-x)); }
__device__ __forceinline__ float sigmoidf_(float x) { return 1.0f / (1.0f + __expf(-x)); }

// ======================================================================
// 0) init accumulators
// ======================================================================
__global__ void k_init(float* __restrict__ hsem, float* __restrict__ combsum,
                       float* __restrict__ dvsum, float* __restrict__ nsum,
                       float* __restrict__ cnt, unsigned* __restrict__ nmax, int N_)
{
    int total = N_ * (CDIM * 3);
    for (int i = blockIdx.x * blockDim.x + threadIdx.x; i < total;
         i += gridDim.x * blockDim.x) {
        combsum[i] = 0.0f;
        if (i < N_ * CDIM) hsem[i] = 0.0f;
        if (i < N_ * 3)    dvsum[i] = 0.0f;
        if (i < N_ * HHEADS) { nsum[i] = 0.0f; nmax[i] = 0x007FFFFFu; } // enc(-inf)
        if (i < N_)        cnt[i] = 0.0f;
    }
}

// ======================================================================
// 1) repack w_xmix [224x224] f32 row-major -> per-lane WMMA-B f16 fragments
//    tile (kt in 0..6, nt in 0..13): 32 lanes x v16h, lane l: n = nt*16+(l&15),
//    b[2v] = W[32*kt + 2v + 16*(l>>4)][n], b[2v+1] = W[.. +1][n]
// ======================================================================
__global__ void k_prepack(const float* __restrict__ wx, v16h* __restrict__ out)
{
    int tid = blockIdx.x * blockDim.x + threadIdx.x;
    if (tid >= 7 * 14 * 32) return;
    int kt   = tid / 448;
    int rem  = tid - kt * 448;
    int nt   = rem >> 5;
    int lane = rem & 31;
    int hi   = lane >> 4;
    int n    = nt * 16 + (lane & 15);
    v16h b;
#pragma unroll
    for (int v = 0; v < 8; ++v) {
        int k = 32 * kt + 2 * v + 16 * hi;
        b[2 * v]     = (_Float16)wx[k * CDIM + n];
        b[2 * v + 1] = (_Float16)wx[(k + 1) * CDIM + n];
    }
    out[tid] = b;
}

// ======================================================================
// 2) per-edge: rbf filter conv + edge MLP + attention logits (celu*cut)
// ======================================================================
__global__ __launch_bounds__(128)
void k_edge(const float* __restrict__ h, const float* __restrict__ x,
            const int* __restrict__ idx_i, const int* __restrict__ idx_j,
            const float* __restrict__ w_in, const float* __restrict__ b_in,
            const float* __restrict__ w_e1, const float* __restrict__ b_e1,
            const float* __restrict__ w_e2, const float* __restrict__ b_e2,
            const float* __restrict__ w_att, const float* __restrict__ b_att,
            float* __restrict__ he, float* __restrict__ logit,
            float* __restrict__ dir, int E_)
{
    int e = blockIdx.x * blockDim.x + threadIdx.x;
    if (e >= E_) return;
    int i = idx_i[e], j = idx_j[e];

    float rx = x[j * 3 + 0] - x[i * 3 + 0];
    float ry = x[j * 3 + 1] - x[i * 3 + 1];
    float rz = x[j * 3 + 2] - x[i * 3 + 2];
    float d  = sqrtf(rx * rx + ry * ry + rz * rz);
    float inv = 1.0f / (d + EPS_F);
    dir[e * 3 + 0] = rx * inv;
    dir[e * 3 + 1] = ry * inv;
    dir[e * 3 + 2] = rz * inv;

    float hi_[FDIM], hj_[FDIM];
#pragma unroll
    for (int k = 0; k < FDIM; ++k) { hi_[k] = h[i * FDIM + k]; hj_[k] = h[j * FDIM + k]; }

    // t1 = b_e1 + edge_in @ w_e1  (edge_in = [h_i, h_j, filt(50), d])
    float t1[FDIM];
#pragma unroll
    for (int f = 0; f < FDIM; ++f) t1[f] = b_e1[f] + d * w_e1[114 * FDIM + f];
#pragma unroll 4
    for (int k = 0; k < FDIM; ++k) {
        float a = hi_[k], b = hj_[k];
#pragma unroll
        for (int f = 0; f < FDIM; ++f)
            t1[f] += a * w_e1[k * FDIM + f] + b * w_e1[(FDIM + k) * FDIM + f];
    }
    const float wdt = CUTOFF_F / (float)(NRBF_C - 1);
    const float iwdt = 1.0f / wdt;
    for (int c = 0; c < NRBF_C; ++c) {
        float lin = b_in[c];
#pragma unroll
        for (int k = 0; k < FDIM; ++k)
            lin += hi_[k] * w_in[k * NRBF_C + c] + hj_[k] * w_in[(FDIM + k) * NRBF_C + c];
        float z = (d - (float)c * wdt) * iwdt;
        float filt = __expf(-0.5f * z * z) * lin;
#pragma unroll
        for (int f = 0; f < FDIM; ++f)
            t1[f] += filt * w_e1[(2 * FDIM + c) * FDIM + f];
    }
#pragma unroll
    for (int f = 0; f < FDIM; ++f) t1[f] = siluf(t1[f]);

    float heo[FDIM];
#pragma unroll
    for (int f = 0; f < FDIM; ++f) heo[f] = b_e2[f];
#pragma unroll 4
    for (int k = 0; k < FDIM; ++k) {
        float a = t1[k];
#pragma unroll
        for (int f = 0; f < FDIM; ++f) heo[f] += a * w_e2[k * FDIM + f];
    }
#pragma unroll
    for (int f = 0; f < FDIM; ++f) he[e * FDIM + f] = heo[f];

    float cut = (d < CUTOFF_F) ? 0.5f * (__cosf(PI_F * d / CUTOFF_F) + 1.0f) : 0.0f;
#pragma unroll
    for (int hh = 0; hh < HHEADS; ++hh) {
        float a = b_att[hh];
#pragma unroll
        for (int f = 0; f < FDIM; ++f) a += heo[f] * w_att[f * HHEADS + hh];
        // celu(a, alpha=2) = max(0,a) + min(0, 2*(exp(a/2)-1))
        float cel = fmaxf(a, 0.0f) + fminf(0.0f, 2.0f * (__expf(fminf(a, 0.0f) * 0.5f) - 1.0f));
        logit[e * HHEADS + hh] = cel * cut;
    }
}

// ======================================================================
// 3a) scatter max (ordered-uint encoding of float)
// ======================================================================
__global__ void k_att_max(const float* __restrict__ logit, const int* __restrict__ idx_i,
                          unsigned* __restrict__ nmax, int E_)
{
    int t = blockIdx.x * blockDim.x + threadIdx.x;
    if (t >= E_ * HHEADS) return;
    int e = t / HHEADS, hh = t - e * HHEADS;
    float l = logit[t];
    unsigned bits = __float_as_uint(l);
    unsigned key = (bits & 0x80000000u) ? ~bits : (bits | 0x80000000u);
    atomicMax(&nmax[idx_i[e] * HHEADS + hh], key);
}

// 3b) exp(l - max), accumulate denominator, count degree
__global__ void k_att_exp(float* __restrict__ logit, const int* __restrict__ idx_i,
                          const unsigned* __restrict__ nmax, float* __restrict__ nsum,
                          float* __restrict__ cnt, int E_)
{
    int e = blockIdx.x * blockDim.x + threadIdx.x;
    if (e >= E_) return;
    int i = idx_i[e];
    atomicAdd(&cnt[i], 1.0f);
#pragma unroll
    for (int hh = 0; hh < HHEADS; ++hh) {
        unsigned k = nmax[i * HHEADS + hh];
        unsigned bits = (k & 0x80000000u) ? (k & 0x7FFFFFFFu) : ~k;
        float m = __uint_as_float(bits);
        if (!(m > -1e30f)) m = 0.0f;                 // guard empty/-inf like reference
        float ev = __expf(logit[e * HHEADS + hh] - m);
        logit[e * HHEADS + hh] = ev;
        atomicAdd(&nsum[i * HHEADS + hh], ev);
    }
}

// ======================================================================
// 4) WMMA kernel: pre_tanh = h_ij_sem @ w_xmix, fused tanh + segment sums
//    block = 16 edges x 14 waves (448 thr); wave w owns output cols [16w,16w+16)
//    h_ij_sem tile is built ONCE in LDS (f16), A-fragments are two
//    contiguous 16B ds_load_b128 spans per K-chunk; K-loop fully unrolled.
// ======================================================================
__global__ __launch_bounds__(448)
void k_gemm(const float* __restrict__ he, const float* __restrict__ expw,
            const float* __restrict__ nsum, const float* __restrict__ dir,
            const int* __restrict__ idx_i, const v16h* __restrict__ wxh,
            const float* __restrict__ wvmix,
            float* __restrict__ hsem, float* __restrict__ combsum,
            float* __restrict__ dvsum, int E_)
{
    __shared__ float s_he[16][33];
    __shared__ float s_att[16][8];
    __shared__ float s_dir[16][4];
    __shared__ float s_sv[16];
    __shared__ int   s_idx[16];
    __shared__ __align__(16) _Float16 s_sem[16][240];   // 224 + pad (16B-aligned rows)

    const int e0 = blockIdx.x * 16;
    const int tid = threadIdx.x;

    // ---- phase 1: stage h_edge, normalized att, dir, idx ----
    for (int p = tid; p < 16 * FDIM; p += 448) {
        int m = p >> 5, f = p & 31;
        int e = e0 + m;
        s_he[m][f] = (e < E_) ? he[e * FDIM + f] : 0.0f;
    }
    for (int p = tid; p < 16 * HHEADS; p += 448) {
        int m = p / HHEADS, hh = p - m * HHEADS;
        int e = e0 + m;
        float a = 0.0f;
        if (e < E_) {
            int i = idx_i[e];
            a = expw[e * HHEADS + hh] / fmaxf(nsum[i * HHEADS + hh], 1e-30f);
        }
        s_att[m][hh] = a;
    }
    for (int p = tid; p < 16 * 3; p += 448) {
        int m = p / 3, xx = p - m * 3;
        int e = e0 + m;
        s_dir[m][xx] = (e < E_) ? dir[e * 3 + xx] : 0.0f;
    }
    if (tid < 16) {
        int e = e0 + tid;
        s_sv[tid] = 0.0f;
        s_idx[tid] = (e < E_) ? idx_i[e] : 0;
    }
    __syncthreads();

    // ---- phase 2: build f16 h_ij_sem tile in LDS, scatter h_i_sem once ----
    for (int p = tid; p < 16 * CDIM; p += 448) {
        int m = p / CDIM, c = p - m * CDIM;
        float val = s_he[m][c / HHEADS] * s_att[m][c % HHEADS];
        s_sem[m][c] = (_Float16)val;
        if (e0 + m < E_)
            atomicAdd(&hsem[s_idx[m] * CDIM + c], val);
    }
    __syncthreads();

    // ---- phase 3: WMMA over K = 224 (7 x 32), fully unrolled ----
    const int wid = tid >> 5, lane = tid & 31;
    const int hi = lane >> 4;
    const int n = wid * 16 + (lane & 15);
    const int arow = lane & 15;                 // A-fragment row for this lane
    const float wv = wvmix[n];

    v8f acc = {0.f, 0.f, 0.f, 0.f, 0.f, 0.f, 0.f, 0.f};
#pragma unroll
    for (int kt = 0; kt < 7; ++kt) {
        v16h b = wxh[(kt * 14 + wid) * 32 + lane];
        v8h alo = *(const v8h*)&s_sem[arow][32 * kt + 16 * hi];
        v8h ahi = *(const v8h*)&s_sem[arow][32 * kt + 16 * hi + 8];
        v16h a = __builtin_shufflevector(alo, ahi,
                                         0, 1, 2, 3, 4, 5, 6, 7,
                                         8, 9, 10, 11, 12, 13, 14, 15);
        acc = __builtin_amdgcn_wmma_f32_16x16x32_f16(false, a, false, b,
                                                     (short)0, acc, false, false);
    }

    // ---- phase 4: tanh + segment scatters ----
#pragma unroll
    for (int v = 0; v < 8; ++v) {
        int m = v + 8 * hi;                     // D-fragment row
        int e = e0 + m;
        if (e < E_) {
            float t = tanhf(acc[v]);
            int i = s_idx[m];
            atomicAdd(&combsum[(i * CDIM + n) * 3 + 0], t * s_dir[m][0]);
            atomicAdd(&combsum[(i * CDIM + n) * 3 + 1], t * s_dir[m][1]);
            atomicAdd(&combsum[(i * CDIM + n) * 3 + 2], t * s_dir[m][2]);
            atomicAdd(&s_sv[m], t * wv);        // v_ij scalar partial (ds_add_f32)
        }
    }
    __syncthreads();
    if (tid < 16) {
        int e = e0 + tid;
        if (e < E_) {
            int i = s_idx[tid];
            float s = s_sv[tid];
            atomicAdd(&dvsum[i * 3 + 0], s * s_dir[tid][0]);
            atomicAdd(&dvsum[i * 3 + 1], s * s_dir[tid][1]);
            atomicAdd(&dvsum[i * 3 + 2], s * s_dir[tid][2]);
        }
    }
}

// ======================================================================
// 5) per-node: spatial MLP, node MLP, velocity gate, outputs
// ======================================================================
__global__ __launch_bounds__(128)
void k_node(const float* __restrict__ h, const float* __restrict__ x,
            const float* __restrict__ v,
            const float* __restrict__ w_n1, const float* __restrict__ b_n1,
            const float* __restrict__ w_n2, const float* __restrict__ b_n2,
            const float* __restrict__ w_pn1, const float* __restrict__ b_pn1,
            const float* __restrict__ w_pn2, const float* __restrict__ b_pn2,
            const float* __restrict__ w_v1, const float* __restrict__ b_v1,
            const float* __restrict__ w_v2,
            const float* __restrict__ hsem, const float* __restrict__ combsum,
            const float* __restrict__ dvsum, const float* __restrict__ cnt,
            float* __restrict__ out_h, float* __restrict__ out_x,
            float* __restrict__ out_v, int N_)
{
    int i = blockIdx.x * blockDim.x + threadIdx.x;
    if (i >= N_) return;

    float hrow[FDIM];
#pragma unroll
    for (int k = 0; k < FDIM; ++k) hrow[k] = h[i * FDIM + k];
    float rin = 1.0f / fmaxf(cnt[i], 1.0f);

    // spatial: cnsq @ w_pn1 -> silu -> @ w_pn2 -> silu
    float p1[FDIM];
#pragma unroll
    for (int f = 0; f < FDIM; ++f) p1[f] = b_pn1[f];
    for (int c = 0; c < CDIM; ++c) {
        float sx = combsum[(i * CDIM + c) * 3 + 0] * rin;
        float sy = combsum[(i * CDIM + c) * 3 + 1] * rin;
        float sz = combsum[(i * CDIM + c) * 3 + 2] * rin;
        float q = sx * sx + sy * sy + sz * sz;
#pragma unroll
        for (int f = 0; f < FDIM; ++f) p1[f] += q * w_pn1[c * FDIM + f];
    }
#pragma unroll
    for (int f = 0; f < FDIM; ++f) p1[f] = siluf(p1[f]);
    float spat[FDIM];
#pragma unroll
    for (int f = 0; f < FDIM; ++f) spat[f] = b_pn2[f];
#pragma unroll 4
    for (int k = 0; k < FDIM; ++k) {
        float a = p1[k];
#pragma unroll
        for (int f = 0; f < FDIM; ++f) spat[f] += a * w_pn2[k * FDIM + f];
    }
#pragma unroll
    for (int f = 0; f < FDIM; ++f) spat[f] = siluf(spat[f]);

    // node MLP: [h, h_i_sem, spatial] @ w_n1 -> silu -> @ w_n2 -> silu
    float n1[FDIM];
#pragma unroll
    for (int f = 0; f < FDIM; ++f) n1[f] = b_n1[f];
#pragma unroll 4
    for (int k = 0; k < FDIM; ++k) {
        float a = hrow[k];
#pragma unroll
        for (int f = 0; f < FDIM; ++f) n1[f] += a * w_n1[k * FDIM + f];
    }
    for (int k = 0; k < CDIM; ++k) {
        float a = hsem[i * CDIM + k];
#pragma unroll
        for (int f = 0; f < FDIM; ++f) n1[f] += a * w_n1[(FDIM + k) * FDIM + f];
    }
#pragma unroll 4
    for (int k = 0; k < FDIM; ++k) {
        float a = spat[k];
#pragma unroll
        for (int f = 0; f < FDIM; ++f) n1[f] += a * w_n1[(FDIM + CDIM + k) * FDIM + f];
    }
#pragma unroll
    for (int f = 0; f < FDIM; ++f) n1[f] = siluf(n1[f]);
    float n2[FDIM];
#pragma unroll
    for (int f = 0; f < FDIM; ++f) n2[f] = b_n2[f];
#pragma unroll 4
    for (int k = 0; k < FDIM; ++k) {
        float a = n1[k];
#pragma unroll
        for (int f = 0; f < FDIM; ++f) n2[f] += a * w_n2[k * FDIM + f];
    }
#pragma unroll
    for (int f = 0; f < FDIM; ++f) out_h[i * FDIM + f] = hrow[f] + siluf(n2[f]);

    // gate = 2*sigmoid(silu(h@w_v1+b_v1)@w_v2)
    float g = 0.0f;
    {
        float v1[FDIM];
#pragma unroll
        for (int f = 0; f < FDIM; ++f) v1[f] = b_v1[f];
#pragma unroll 4
        for (int k = 0; k < FDIM; ++k) {
            float a = hrow[k];
#pragma unroll
            for (int f = 0; f < FDIM; ++f) v1[f] += a * w_v1[k * FDIM + f];
        }
#pragma unroll
        for (int f = 0; f < FDIM; ++f) g += siluf(v1[f]) * w_v2[f];
    }
    float gate = 2.0f * sigmoidf_(g);
#pragma unroll
    for (int xx = 0; xx < 3; ++xx) {
        float dv = dvsum[i * 3 + xx] * rin;
        float vu = gate * v[i * 3 + xx] + dv;
        out_v[i * 3 + xx] = vu;
        out_x[i * 3 + xx] = x[i * 3 + xx] + vu;
    }
}

// ======================================================================
// launch
// ======================================================================
extern "C" void kernel_launch(void* const* d_in, const int* in_sizes, int n_in,
                              void* d_out, int out_size, void* d_ws, size_t ws_size,
                              hipStream_t stream)
{
    const float* h     = (const float*)d_in[0];
    const float* x     = (const float*)d_in[1];
    const float* v     = (const float*)d_in[2];
    const int*   pair  = (const int*)d_in[3];
    const float* w_in  = (const float*)d_in[4];
    const float* b_in  = (const float*)d_in[5];
    const float* w_e1  = (const float*)d_in[6];
    const float* b_e1  = (const float*)d_in[7];
    const float* w_e2  = (const float*)d_in[8];
    const float* b_e2  = (const float*)d_in[9];
    const float* w_att = (const float*)d_in[10];
    const float* b_att = (const float*)d_in[11];
    const float* w_n1  = (const float*)d_in[12];
    const float* b_n1  = (const float*)d_in[13];
    const float* w_n2  = (const float*)d_in[14];
    const float* b_n2  = (const float*)d_in[15];
    const float* w_pn1 = (const float*)d_in[16];
    const float* b_pn1 = (const float*)d_in[17];
    const float* w_pn2 = (const float*)d_in[18];
    const float* b_pn2 = (const float*)d_in[19];
    const float* w_v1  = (const float*)d_in[20];
    const float* b_v1  = (const float*)d_in[21];
    const float* w_v2  = (const float*)d_in[22];
    const float* w_xmix = (const float*)d_in[23];
    const float* w_vmix = (const float*)d_in[24];

    const int N_ = in_sizes[0] / FDIM;
    const int E_ = in_sizes[3] / 2;
    const int* idx_i = pair;
    const int* idx_j = pair + E_;

    // workspace bump allocator (256B aligned)
    char* wp = (char*)d_ws;
    auto alloc = [&](size_t bytes) -> void* {
        void* p = (void*)wp;
        wp += (bytes + 255) & ~(size_t)255;
        return p;
    };
    float*    he      = (float*)alloc((size_t)E_ * FDIM * 4);
    float*    logit   = (float*)alloc((size_t)E_ * HHEADS * 4);
    float*    dir     = (float*)alloc((size_t)E_ * 3 * 4);
    unsigned* nmax    = (unsigned*)alloc((size_t)N_ * HHEADS * 4);
    float*    nsum    = (float*)alloc((size_t)N_ * HHEADS * 4);
    float*    cnt     = (float*)alloc((size_t)N_ * 4);
    float*    hsem    = (float*)alloc((size_t)N_ * CDIM * 4);
    float*    combsum = (float*)alloc((size_t)N_ * CDIM * 3 * 4);
    float*    dvsum   = (float*)alloc((size_t)N_ * 3 * 4);
    v16h*     wxh     = (v16h*)alloc((size_t)CDIM * CDIM * 2);

    float* out_h = (float*)d_out;
    float* out_x = out_h + (size_t)N_ * FDIM;
    float* out_v = out_x + (size_t)N_ * 3;

    {
        int total = N_ * CDIM * 3;
        k_init<<<(total + 255) / 256, 256, 0, stream>>>(hsem, combsum, dvsum, nsum, cnt, nmax, N_);
    }
    k_prepack<<<(7 * 14 * 32 + 255) / 256, 256, 0, stream>>>(w_xmix, wxh);
    k_edge<<<(E_ + 127) / 128, 128, 0, stream>>>(h, x, idx_i, idx_j, w_in, b_in,
                                                 w_e1, b_e1, w_e2, b_e2, w_att, b_att,
                                                 he, logit, dir, E_);
    k_att_max<<<((size_t)E_ * HHEADS + 255) / 256, 256, 0, stream>>>(logit, idx_i, nmax, E_);
    k_att_exp<<<(E_ + 255) / 256, 256, 0, stream>>>(logit, idx_i, nmax, nsum, cnt, E_);
    k_gemm<<<(E_ + 15) / 16, 448, 0, stream>>>(he, logit, nsum, dir, idx_i, wxh, w_vmix,
                                               hsem, combsum, dvsum, E_);
    k_node<<<(N_ + 127) / 128, 128, 0, stream>>>(h, x, v, w_n1, b_n1, w_n2, b_n2,
                                                 w_pn1, b_pn1, w_pn2, b_pn2,
                                                 w_v1, b_v1, w_v2,
                                                 hsem, combsum, dvsum, cnt,
                                                 out_h, out_x, out_v, N_);
}
